// SelfAttention_223338299920
// MI455X (gfx1250) — compile-verified, matrix operand order checked by reading, not run on hardware
//
#include <hip/hip_runtime.h>
#include <hip/hip_bf16.h>

// ---------------------------------------------------------------------------
// Self-attention (B=4, N=2048, D=1024, H=16, HD=64) for MI455X (gfx1250).
// bf16 v_wmma_f32_16x16x32_bf16 everywhere (fp32 accum); flash attention so
// scores never leave the WGP; K/V tiles staged into LDS by the Tensor Data
// Mover (one wave issues, whole block consumes); ping-pong double-buffered
// GEMMs so loads overlap the matrix pipe with no register-rotation copies.
// wave32, 8 waves / 256-thread block.
// ---------------------------------------------------------------------------

typedef __bf16 bf16;
typedef __attribute__((ext_vector_type(16))) __bf16 bf16x16;
typedef __attribute__((ext_vector_type(4)))  __bf16 bf16x4;
typedef __attribute__((ext_vector_type(8)))  float  v8f;
typedef __attribute__((ext_vector_type(4)))  float  f32x4;

#define B_  4
#define N_  2048
#define D_  1024
#define H_  16
#define HD_ 64
#define SCALE_ 0.03125f   // D^-0.5 = 1/32 (reference scales by D, not HD)

#if defined(__has_builtin)
#  if __has_builtin(__builtin_amdgcn_tensor_load_to_lds)
#    define USE_TDM 1
#  endif
#endif
#ifndef USE_TDM
#  define USE_TDM 0
#endif

union Frag16 { bf16x16 v; f32x4 h[2]; };

// A-style 16x32 bf16 fragment load (ISA 7.12.2): lane L -> row M = L%16,
// half = L/16; VGPR0..3 = K k0+half*8..+7, VGPR4..7 = K k0+16+half*8..+7
// => two B128 loads per lane. B-fragments of a transposed operand (NxK)
// stripe identically, so rows of B^T load the same way.
__device__ __forceinline__ bf16x16 load_frag(const bf16* p) {
    Frag16 f;
    f.h[0] = *(const f32x4*)(p);
    f.h[1] = *(const f32x4*)(p + 16);
    return f.v;
}

__device__ __forceinline__ v8f wmma_bf16(bf16x16 a, bf16x16 b, v8f c) {
    return __builtin_amdgcn_wmma_f32_16x16x32_bf16(
        false, a, false, b, (short)0, c, false, false);
}

#if USE_TDM
typedef __attribute__((ext_vector_type(4))) unsigned int u32x4;
typedef __attribute__((ext_vector_type(8))) int          i32x8;
typedef __attribute__((ext_vector_type(4))) int          i32x4;

// One TDM 2D tile load: global (row-major, stride0 elements of 2B) -> LDS
// (rows packed contiguously). Descriptor per ISA 08_async_tensor §8:
//   group0: [1:0]=count=1, [63:32]=lds_addr, [120:64]=global_addr, type=2
//   group1: data_size=2B, tensor dims huge (tiles never clip), tile_d0 x
//   tile_d1, tensor_dim0_stride = stride0. Higher groups zero (2D tile).
// This toolchain's builtin: (u32x4, i32x8, i32x4, i32x4, i32x8, i32 cpol).
__device__ __forceinline__ void tdm_load_2d(unsigned lds_addr, const void* gptr,
                                            unsigned tile_d0, unsigned tile_d1,
                                            unsigned stride0) {
    unsigned long long ga = (unsigned long long)(uintptr_t)gptr;
    u32x4 g0;
    g0[0] = 1u;                                          // count=1, user D#
    g0[1] = lds_addr;                                    // LDS byte address
    g0[2] = (unsigned)(ga & 0xFFFFFFFFu);                // global_addr[31:0]
    g0[3] = (unsigned)((ga >> 32) & 0x1FFFFFFu) | (2u << 30);  // [56:32]|type=2
    const unsigned td0 = 1u << 20, td1 = 1u << 20;       // generous tensor dims
    i32x8 g1;
    g1[0] = (int)(1u << 16);                             // data_size = 2 bytes
    g1[1] = (int)((td0 & 0xFFFFu) << 16);                // tensor_dim0 lo16
    g1[2] = (int)((td0 >> 16) | ((td1 & 0xFFFFu) << 16));
    g1[3] = (int)((td1 >> 16) | (tile_d0 << 16));        // | tile_dim0
    g1[4] = (int)tile_d1;                                // tile_dim1, dim2=0
    g1[5] = (int)stride0;                                // tensor_dim0_stride
    g1[6] = 0;
    g1[7] = 0;
    i32x4 z4 = {0, 0, 0, 0};
    i32x8 z8 = {0, 0, 0, 0, 0, 0, 0, 0};
    __builtin_amdgcn_tensor_load_to_lds(g0, g1, z4, z4, z8, 0);
}
#endif

// ---------------------------------------------------------------------------
// fp32 -> bf16 elementwise convert (4 elems / lane)
// ---------------------------------------------------------------------------
__global__ void __launch_bounds__(256) k_cvt4(const f32x4* __restrict__ x,
                                              bf16* __restrict__ y, int n4) {
    int i = blockIdx.x * 256 + threadIdx.x;
    if (i < n4) {
        f32x4 f = x[i];
        bf16x4 o;
        o.x = (bf16)f.x; o.y = (bf16)f.y; o.z = (bf16)f.z; o.w = (bf16)f.w;
        *(bf16x4*)(y + 4 * (size_t)i) = o;
    }
}

// ---------------------------------------------------------------------------
// Transpose + convert: w [K, Ncols] fp32 -> wt [Ncols, K] bf16 (one-time).
// ---------------------------------------------------------------------------
__global__ void __launch_bounds__(256) k_transpose_cvt(const float* __restrict__ w,
                                                       bf16* __restrict__ wt,
                                                       int K, int Ncols) {
    int i = blockIdx.x * 256 + threadIdx.x;
    if (i < K * Ncols) {
        int n = i % Ncols;
        int k = i / Ncols;
        wt[(size_t)n * K + k] = (bf16)w[i];
    }
}

// ---------------------------------------------------------------------------
// QKV GEMM: C[8192,3072] = xb @ wqkvT^T. Ping-pong k-loop (step 64 = two
// WMMA k-steps): even/odd fragment sets keep fixed registers, each set's
// loads issue while the other set feeds the wmmas -> partial loadcnt waits,
// zero rotation movs. Epilogue scatters q,k -> [B,H,N,64]; v -> [B,H,64,N].
// ---------------------------------------------------------------------------
__global__ void __launch_bounds__(256) k_gemm_qkv(const bf16* __restrict__ A,
                                                  const bf16* __restrict__ Bt,
                                                  bf16* __restrict__ q,
                                                  bf16* __restrict__ k,
                                                  bf16* __restrict__ vT) {
    const int lane = threadIdx.x & 31, wave = threadIdx.x >> 5;
    const int m16 = lane & 15, half = lane >> 4;
    const int row0 = blockIdx.x * 128 + wave * 16;
    const int col0 = blockIdx.y * 64;

    const bf16* ap = A + (size_t)(row0 + m16) * D_ + half * 8;
    const bf16* bp = Bt + (size_t)(col0 + m16) * D_ + half * 8;

    v8f acc[4] = {};
    bf16x16 aE = load_frag(ap);
    bf16x16 bE[4];
#pragma unroll
    for (int t = 0; t < 4; ++t) bE[t] = load_frag(bp + (size_t)t * 16 * D_);
    bf16x16 aO;
    bf16x16 bO[4];

    for (int kk = 0; kk < D_ - 64; kk += 64) {
        aO = load_frag(ap + kk + 32);
#pragma unroll
        for (int t = 0; t < 4; ++t) bO[t] = load_frag(bp + (size_t)t * 16 * D_ + kk + 32);
#pragma unroll
        for (int t = 0; t < 4; ++t) acc[t] = wmma_bf16(aE, bE[t], acc[t]);
        aE = load_frag(ap + kk + 64);
#pragma unroll
        for (int t = 0; t < 4; ++t) bE[t] = load_frag(bp + (size_t)t * 16 * D_ + kk + 64);
#pragma unroll
        for (int t = 0; t < 4; ++t) acc[t] = wmma_bf16(aO, bO[t], acc[t]);
    }
    aO = load_frag(ap + D_ - 32);
#pragma unroll
    for (int t = 0; t < 4; ++t) bO[t] = load_frag(bp + (size_t)t * 16 * D_ + D_ - 32);
#pragma unroll
    for (int t = 0; t < 4; ++t) acc[t] = wmma_bf16(aE, bE[t], acc[t]);
#pragma unroll
    for (int t = 0; t < 4; ++t) acc[t] = wmma_bf16(aO, bO[t], acc[t]);

    // C/D layout: lane -> col n = lane%16 ; rows m = half*8 + v
#pragma unroll
    for (int t = 0; t < 4; ++t) {
        int nc  = col0 + t * 16 + m16;
        int sec = nc >> 10;                      // 0=q 1=k 2=v
        int d   = nc & (D_ - 1);
        int h   = d >> 6;
        int hd  = d & 63;
#pragma unroll
        for (int v = 0; v < 8; ++v) {
            int r  = row0 + half * 8 + v;
            int bq = r >> 11;
            int nq = r & (N_ - 1);
            bf16 val = (bf16)acc[t][v];
            if (sec == 0)
                q[(((size_t)(bq * H_ + h) * N_ + nq) << 6) + hd] = val;
            else if (sec == 1)
                k[(((size_t)(bq * H_ + h) * N_ + nq) << 6) + hd] = val;
            else
                vT[(((size_t)(bq * H_ + h) * HD_ + hd) << 11) + nq] = val;
        }
    }
}

// ---------------------------------------------------------------------------
// Flash attention. grid = (B*H, N/128); each wave owns 16 query rows.
// TDM path: wave 0 DMA's the shared 32-key K tile (32x64) and V tile (64x32)
// into double-buffered LDS; all 8 waves consume via ds_load_b128 fragments.
// Softmax is an in-register online update with shfl_xor row reductions; the
// P tile bounces through 1KB/wave LDS to re-stripe C-layout -> A-fragment.
// ---------------------------------------------------------------------------
__global__ void __launch_bounds__(256) k_flash(const bf16* __restrict__ Q,
                                               const bf16* __restrict__ K,
                                               const bf16* __restrict__ Vt,
                                               bf16* __restrict__ O) {
#if USE_TDM
    __shared__ bf16 sK[2][32 * HD_];            // 2 x 4KB, shared by block
    __shared__ bf16 sV[2][HD_ * 32];            // 2 x 4KB, shared by block
#endif
    __shared__ bf16 sP[8][16 * 32];             // per-wave P staging (8KB)

    const int lane = threadIdx.x & 31, wave = threadIdx.x >> 5;
    const int m16 = lane & 15, half = lane >> 4;
    const int bh = blockIdx.x;                  // b*16 + h
    const int q0row = blockIdx.y * 128 + wave * 16;

    const bf16* Qp = Q  + ((size_t)bh * N_ + q0row) * HD_;
    const bf16* Kp = K  + (size_t)bh * N_ * HD_;
    const bf16* Vp = Vt + (size_t)bh * HD_ * N_;

    bf16x16 qa0 = load_frag(Qp + (size_t)m16 * HD_ + half * 8);
    bf16x16 qa1 = load_frag(Qp + (size_t)m16 * HD_ + 32 + half * 8);

    float mrow[8], lrow[8];
#pragma unroll
    for (int v = 0; v < 8; ++v) { mrow[v] = -1e30f; lrow[v] = 0.0f; }
    v8f acc[4] = {};

#if USE_TDM
    if (wave == 0) {                            // prime buffer 0
        tdm_load_2d((unsigned)(uintptr_t)&sK[0][0], Kp, HD_, 32, HD_);
        tdm_load_2d((unsigned)(uintptr_t)&sV[0][0], Vp, 32, HD_, N_);
        __builtin_amdgcn_s_wait_tensorcnt(0);
    }
    __syncthreads();
#endif

    for (int kt = 0; kt < N_ / 32; ++kt) {
        const int j0 = kt * 32;
#if USE_TDM
        const int cur = kt & 1, nxt = cur ^ 1;
        if (wave == 0 && kt + 1 < N_ / 32) {    // DMA next tiles during compute
            tdm_load_2d((unsigned)(uintptr_t)&sK[nxt][0],
                        Kp + (size_t)(j0 + 32) * HD_, HD_, 32, HD_);
            tdm_load_2d((unsigned)(uintptr_t)&sV[nxt][0],
                        Vp + (j0 + 32), 32, HD_, N_);
        }
#endif
        // ---- scores for two 16-key sub-tiles -------------------------------
        v8f s[2];
#pragma unroll
        for (int jj = 0; jj < 2; ++jj) {
#if USE_TDM
            const bf16* kp = &sK[cur][(jj * 16 + m16) * HD_ + half * 8];
#else
            const bf16* kp = Kp + (size_t)(j0 + jj * 16 + m16) * HD_ + half * 8;
#endif
            bf16x16 kb0 = load_frag(kp);
            bf16x16 kb1 = load_frag(kp + 32);
            v8f t = {};
            t = wmma_bf16(qa0, kb0, t);
            t = wmma_bf16(qa1, kb1, t);
            s[jj] = t;
        }
        // ---- V fragments early so loads overlap softmax VALU ---------------
        Frag16 vb[4];
#pragma unroll
        for (int t = 0; t < 4; ++t) {
#if USE_TDM
            const bf16* vp = &sV[cur][(t * 16 + m16) * 32 + half * 8];
            vb[t].h[0] = *(const f32x4*)vp;
            vb[t].h[1] = *(const f32x4*)(vp + 16);
#else
            const bf16* vp = Vp + (size_t)(t * 16 + m16) * N_ + j0 + half * 8;
            vb[t].h[0] = *(const f32x4*)vp;
            vb[t].h[1] = *(const f32x4*)(vp + 16);
#endif
        }
        // ---- online softmax over the 32 new keys ---------------------------
        float corr[8];
#pragma unroll
        for (int v = 0; v < 8; ++v) {
            float a = s[0][v] * SCALE_;
            float b = s[1][v] * SCALE_;
            float r = fmaxf(a, b);
            r = fmaxf(r, __shfl_xor(r, 1));
            r = fmaxf(r, __shfl_xor(r, 2));
            r = fmaxf(r, __shfl_xor(r, 4));
            r = fmaxf(r, __shfl_xor(r, 8));     // row max in 16-lane group
            float mn = fmaxf(mrow[v], r);
            float p0 = __expf(a - mn);
            float p1 = __expf(b - mn);
            s[0][v] = p0; s[1][v] = p1;
            float ps = p0 + p1;
            ps += __shfl_xor(ps, 1);
            ps += __shfl_xor(ps, 2);
            ps += __shfl_xor(ps, 4);
            ps += __shfl_xor(ps, 8);            // row sum
            float c = __expf(mrow[v] - mn);
            corr[v] = c;
            lrow[v] = lrow[v] * c + ps;
            mrow[v] = mn;
        }
#pragma unroll
        for (int t = 0; t < 4; ++t)
#pragma unroll
            for (int v = 0; v < 8; ++v) acc[t][v] *= corr[v];

        // ---- re-stripe P (C-layout -> A-fragment) via per-wave LDS ---------
#pragma unroll
        for (int jj = 0; jj < 2; ++jj)
#pragma unroll
            for (int v = 0; v < 8; ++v)
                sP[wave][(half * 8 + v) * 32 + jj * 16 + m16] = (bf16)s[jj][v];
        asm volatile("s_wait_dscnt 0" ::: "memory");  // cross-lane, same wave

        Frag16 pa;
        pa.h[0] = *(const f32x4*)&sP[wave][m16 * 32 + half * 8];
        pa.h[1] = *(const f32x4*)&sP[wave][m16 * 32 + 16 + half * 8];

        // ---- P . V ---------------------------------------------------------
#pragma unroll
        for (int t = 0; t < 4; ++t)
            acc[t] = wmma_bf16(pa.v, vb[t].v, acc[t]);
        asm volatile("s_wait_dscnt 0" ::: "memory");  // WAR guard on sP

#if USE_TDM
        if (wave == 0) __builtin_amdgcn_s_wait_tensorcnt(0);
        __syncthreads();                        // publish next K/V buffers
#endif
    }

    // ---- normalize + write attn-out [B*N, D] bf16 --------------------------
    const int b = bh >> 4, h = bh & 15;
#pragma unroll
    for (int t = 0; t < 4; ++t) {
#pragma unroll
        for (int v = 0; v < 8; ++v) {
            float o = acc[t][v] / lrow[v];
            int r = q0row + half * 8 + v;
            O[((size_t)(b * N_ + r)) * D_ + h * HD_ + t * 16 + m16] = (bf16)o;
        }
    }
}

// ---------------------------------------------------------------------------
// Output projection: out[8192,1024] = ao @ wprojT^T + bias (fp32 out),
// same ping-pong pipelined k-loop as k_gemm_qkv.
// ---------------------------------------------------------------------------
__global__ void __launch_bounds__(256) k_gemm_proj(const bf16* __restrict__ A,
                                                   const bf16* __restrict__ Bt,
                                                   const float* __restrict__ bias,
                                                   float* __restrict__ C) {
    const int lane = threadIdx.x & 31, wave = threadIdx.x >> 5;
    const int m16 = lane & 15, half = lane >> 4;
    const int row0 = blockIdx.x * 128 + wave * 16;
    const int col0 = blockIdx.y * 64;

    const bf16* ap = A + (size_t)(row0 + m16) * D_ + half * 8;
    const bf16* bp = Bt + (size_t)(col0 + m16) * D_ + half * 8;

    v8f acc[4] = {};
    bf16x16 aE = load_frag(ap);
    bf16x16 bE[4];
#pragma unroll
    for (int t = 0; t < 4; ++t) bE[t] = load_frag(bp + (size_t)t * 16 * D_);
    bf16x16 aO;
    bf16x16 bO[4];

    for (int kk = 0; kk < D_ - 64; kk += 64) {
        aO = load_frag(ap + kk + 32);
#pragma unroll
        for (int t = 0; t < 4; ++t) bO[t] = load_frag(bp + (size_t)t * 16 * D_ + kk + 32);
#pragma unroll
        for (int t = 0; t < 4; ++t) acc[t] = wmma_bf16(aE, bE[t], acc[t]);
        aE = load_frag(ap + kk + 64);
#pragma unroll
        for (int t = 0; t < 4; ++t) bE[t] = load_frag(bp + (size_t)t * 16 * D_ + kk + 64);
#pragma unroll
        for (int t = 0; t < 4; ++t) acc[t] = wmma_bf16(aO, bO[t], acc[t]);
    }
    aO = load_frag(ap + D_ - 32);
#pragma unroll
    for (int t = 0; t < 4; ++t) bO[t] = load_frag(bp + (size_t)t * 16 * D_ + D_ - 32);
#pragma unroll
    for (int t = 0; t < 4; ++t) acc[t] = wmma_bf16(aE, bE[t], acc[t]);
#pragma unroll
    for (int t = 0; t < 4; ++t) acc[t] = wmma_bf16(aO, bO[t], acc[t]);

#pragma unroll
    for (int t = 0; t < 4; ++t) {
        int nc = col0 + t * 16 + m16;
        float bv = bias[nc];
#pragma unroll
        for (int v = 0; v < 8; ++v) {
            int r = row0 + half * 8 + v;
            C[(size_t)r * D_ + nc] = acc[t][v] + bv;
        }
    }
}

// ---------------------------------------------------------------------------
// Host launcher. Workspace layout (bytes):
//   [0,16M)  xb     bf16 [8192,1024]      [16,22M) wqkvT bf16 [3072,1024]
//   [22,24M) wprojT bf16 [1024,1024]      [24,40M) q     bf16 [B,H,N,64]
//   [40,56M) k      bf16 [B,H,N,64]       [56,72M) vT    bf16 [B,H,64,N]
//   [72,88M) ao     bf16 [8192,1024]
// ---------------------------------------------------------------------------
extern "C" void kernel_launch(void* const* d_in, const int* in_sizes, int n_in,
                              void* d_out, int out_size, void* d_ws, size_t ws_size,
                              hipStream_t stream) {
    const float* x      = (const float*)d_in[0];
    const float* w_qkv  = (const float*)d_in[1];
    const float* w_proj = (const float*)d_in[2];
    const float* b_proj = (const float*)d_in[3];
    float*       out    = (float*)d_out;

    char* ws = (char*)d_ws;
    const size_t MB = 1u << 20;
    bf16* xb     = (bf16*)(ws + 0 * MB);
    bf16* wqkvT  = (bf16*)(ws + 16 * MB);
    bf16* wprojT = (bf16*)(ws + 22 * MB);
    bf16* qb     = (bf16*)(ws + 24 * MB);
    bf16* kb     = (bf16*)(ws + 40 * MB);
    bf16* vT     = (bf16*)(ws + 56 * MB);
    bf16* ao     = (bf16*)(ws + 72 * MB);

    const int nX = B_ * N_ * D_;
    k_cvt4<<<nX / 4 / 256, 256, 0, stream>>>((const f32x4*)x, xb, nX / 4);
    k_transpose_cvt<<<(D_ * 3 * D_) / 256, 256, 0, stream>>>(w_qkv, wqkvT, D_, 3 * D_);
    k_transpose_cvt<<<(D_ * D_) / 256, 256, 0, stream>>>(w_proj, wprojT, D_, D_);

    k_gemm_qkv<<<dim3((B_ * N_) / 128, (3 * D_) / 64), 256, 0, stream>>>(xb, wqkvT, qb, kb, vT);
    k_flash<<<dim3(B_ * H_, N_ / 128), 256, 0, stream>>>(qb, kb, vT, ao);
    k_gemm_proj<<<dim3((B_ * N_) / 128, D_ / 64), 256, 0, stream>>>(ao, wprojT, b_proj, out);
}
